// GNNConditioner_36258113913103
// MI455X (gfx1250) — matrix operand order repaired and checked by reading, not compiled, not run on hardware
//
#include <hip/hip_runtime.h>

// ---------------------------------------------------------------------------
// Model constants (from the reference)
// ---------------------------------------------------------------------------
#define BSZ     4096
#define NA      32
#define NF      64
#define NRBF    16
#define NMSG    32
#define DREST   160
#define KFEAT   2208      // 160 + 32*64, divisible by 32 (69 k-tiles)
#define HIDN    1024
#define DOUT    256
#define CUTR    1.5f

typedef __bf16 v16bf __attribute__((ext_vector_type(16)));
typedef __bf16 v8bf  __attribute__((ext_vector_type(8)));
typedef float  v8f   __attribute__((ext_vector_type(8)));

// ---------------------------------------------------------------------------
// Pack fp32 weight W[K x N] (row-major) into bf16 WMMA B-fragment layout:
//   tile (kt, nt) of 32(K) x 16(N); within tile: lane l -> N = l%16, half=l/16,
//   element ei -> K = half*16 + ei   (contiguous k order, same as the A-fragment loads)
// ---------------------------------------------------------------------------
__global__ void pack_b_kernel(const float* __restrict__ W, __bf16* __restrict__ dst,
                              int K, int N) {
    int idx = blockIdx.x * 256 + threadIdx.x;
    int total = K * N;
    if (idx >= total) return;
    int tile = idx >> 9;
    int rem  = idx & 511;
    int lane = rem >> 4;
    int ei   = rem & 15;
    int Ntiles = N >> 4;
    int kt = tile / Ntiles;
    int nt = tile - kt * Ntiles;
    int n  = nt * 16 + (lane & 15);
    int k  = kt * 32 + (lane >> 4) * 16 + ei;
    dst[idx] = (__bf16)W[(size_t)k * N + n];
}

// ---------------------------------------------------------------------------
// Generic bf16 WMMA GEMM:  C = act(A[M x K] * Bpack + bias)
//   block = 256 threads = 8 waves (2 M-waves x 4 N-waves), block tile 128x128
//   wave tile 64(M) x 32(N) -> 8 accumulators of v_wmma_f32_16x16x32_bf16
//   mode 0: tanh -> bf16 row-major   |   mode 1: identity -> fp32 row-major
// ---------------------------------------------------------------------------
__global__ __launch_bounds__(256, 2)
void gemm_bf16_kernel(const __bf16* __restrict__ A,
                      const __bf16* __restrict__ Bp,
                      const float*  __restrict__ bias,
                      void* __restrict__ Cout,
                      int K, int N, int mode) {
    const int tid  = threadIdx.x;
    const int wid  = tid >> 5;
    const int lane = tid & 31;
    const int wm   = wid & 1;          // 0..1
    const int wn   = wid >> 1;         // 0..3
    const int half = lane >> 4;
    const int l16  = lane & 15;

    const int rowbase = blockIdx.x * 128 + wm * 64;
    const int ntile0  = blockIdx.y * 8 + wn * 2;   // global 16-wide n-tile
    const int Ntiles  = N >> 4;
    const int Ktiles  = K >> 5;

    v8f acc[4][2];
#pragma unroll
    for (int mi = 0; mi < 4; ++mi)
#pragma unroll
        for (int ni = 0; ni < 2; ++ni) acc[mi][ni] = (v8f)(0.0f);

    for (int kt = 0; kt < Ktiles; ++kt) {
        v16bf bfrag[2];
#pragma unroll
        for (int ni = 0; ni < 2; ++ni) {
            const __bf16* p = Bp + ((size_t)(kt * Ntiles + ntile0 + ni) << 9) + lane * 16;
            bfrag[ni] = *(const v16bf*)p;           // coalesced 32B/lane
        }
        v16bf afrag[4];
#pragma unroll
        for (int mi = 0; mi < 4; ++mi) {
            const int row = rowbase + mi * 16 + l16;
            const __bf16* pa = A + (size_t)row * K + kt * 32 + half * 16;
            v8bf lo = *(const v8bf*)pa;             // ei 0..7 : K = half*16 + w
            v8bf hi = *(const v8bf*)(pa + 8);       // ei 8..15: K = half*16 + 8 + w
            v16bf av;
#pragma unroll
            for (int e = 0; e < 8; ++e) { av[e] = lo[e]; av[e + 8] = hi[e]; }
            afrag[mi] = av;
        }
#pragma unroll
        for (int mi = 0; mi < 4; ++mi)
#pragma unroll
            for (int ni = 0; ni < 2; ++ni)
                acc[mi][ni] = __builtin_amdgcn_wmma_f32_16x16x32_bf16(
                    false, afrag[mi], false, bfrag[ni],
                    (short)0, acc[mi][ni], false, false);
    }

    // Epilogue: C fragment element r -> (M = r + 8*half, N = l16)
#pragma unroll
    for (int mi = 0; mi < 4; ++mi) {
#pragma unroll
        for (int ni = 0; ni < 2; ++ni) {
            const int col = (ntile0 + ni) * 16 + l16;
            const float bv = bias[col];
#pragma unroll
            for (int r = 0; r < 8; ++r) {
                const int row = rowbase + mi * 16 + r + half * 8;
                const float cv = acc[mi][ni][r] + bv;
                if (mode == 0)
                    ((__bf16*)Cout)[(size_t)row * N + col] = (__bf16)tanhf(cv);
                else
                    ((float*)Cout)[(size_t)row * N + col] = cv;
            }
        }
    }
}

// ---------------------------------------------------------------------------
// Per-sample GNN front-end: one block (8 waves) per batch sample.
// Produces feat[b, 0:160] = x[b,96:256], feat[b, 160: ] = h3 flattened (bf16).
// ---------------------------------------------------------------------------
__global__ __launch_bounds__(256, 1)
void gnn_kernel(const float* __restrict__ x,
                const float* __restrict__ atom_embed,
                const float* __restrict__ Wm0, const float* __restrict__ bm0,
                const float* __restrict__ Wu0, const float* __restrict__ bu0,
                const float* __restrict__ Wm1, const float* __restrict__ bm1,
                const float* __restrict__ Wu1, const float* __restrict__ bu1,
                const float* __restrict__ Wqkv, const float* __restrict__ bqkv,
                const float* __restrict__ Wq, const float* __restrict__ Wk,
                const float* __restrict__ Wv, const float* __restrict__ Wo,
                const float* __restrict__ bq, const float* __restrict__ bk,
                const float* __restrict__ bv, const float* __restrict__ bo,
                const float* __restrict__ W1f, const float* __restrict__ b1f,
                const float* __restrict__ W2f, const float* __restrict__ b2f,
                const float* __restrict__ g1, const float* __restrict__ be1,
                const float* __restrict__ g2, const float* __restrict__ be2,
                __bf16* __restrict__ feat) {
    __shared__ float spos[NA][3];
    __shared__ float sh[NA][NF];         // running h (also final h2+ff pre-LN)
    __shared__ float sd[NA][NA];         // pairwise distances
    __shared__ float shWm[NA][NMSG];     // h @ Wm[:64]
    __shared__ float sagg[NA][NMSG];     // masked-aggregated messages
    __shared__ float sh2[NA][NF];        // residual / LN buffer
    __shared__ union {
        _Float16 E[NA][NA][NRBF];                          // 32 KB, alive in MP layers
        struct { _Float16 q[NA][NF], k[NA][NF], v[NA][NF], o[NA][NF]; } att; // 16 KB
    } uA;
    __shared__ _Float16 tmp192[NA][192]; // qkv pre-proj; reused as FF hidden
    __shared__ float red[NA][2];         // LN mean / inv-std

    const int tid = threadIdx.x;
    const float* xr = x + (size_t)blockIdx.x * 256;

    if (tid < 96) spos[tid / 3][tid % 3] = xr[tid];
    for (int idx = tid; idx < NA * NF; idx += 256)
        sh[idx >> 6][idx & 63] = atom_embed[idx];
    __syncthreads();

    for (int idx = tid; idx < NA * NA; idx += 256) {
        int i = idx >> 5, j = idx & 31;
        float dx = spos[i][0] - spos[j][0];
        float dy = spos[i][1] - spos[j][1];
        float dz = spos[i][2] - spos[j][2];
        sd[i][j] = sqrtf(fmaxf(dx * dx + dy * dy + dz * dz, 1e-12f));
    }
    __syncthreads();

    // RBF basis, computed once (shared by both MP layers)
    {
        const float w = CUTR / (float)NRBF;
        const float inv2w2 = 1.0f / (2.0f * w * w);
        for (int idx = tid; idx < NA * NA * NRBF; idx += 256) {
            int r = idx & 15, j = (idx >> 4) & 31, i = idx >> 9;
            float c = CUTR * (float)r / 15.0f;   // linspace(0, CUT, 16)
            float t = sd[i][j] - c;
            uA.E[i][j][r] = (_Float16)__expf(-t * t * inv2w2);
        }
    }
    __syncthreads();

    // -------------------- two message-passing layers --------------------
    for (int layer = 0; layer < 2; ++layer) {
        const float* Wm = layer ? Wm1 : Wm0;
        const float* bm = layer ? bm1 : bm0;
        const float* Wu = layer ? Wu1 : Wu0;
        const float* bu = layer ? bu1 : bu0;

        // hWm[j][m] = sum_f h[j][f] * Wm[f][m]   (factored hj part)
        for (int idx = tid; idx < NA * NMSG; idx += 256) {
            int j = idx >> 5, m = idx & 31;
            float s = 0.f;
#pragma unroll 8
            for (int f = 0; f < NF; ++f) s += sh[j][f] * Wm[f * NMSG + m];
            shWm[j][m] = s;
        }
        __syncthreads();

        // agg[i][m] = sum_j mask(i,j) * silu(hWm[j][m] + E[i][j]·WmE[:,m] + bm[m])
        {
            const int m = tid & 31;                 // m == lane -> i wave-uniform
            float wme[NRBF];
#pragma unroll
            for (int r = 0; r < NRBF; ++r) wme[r] = Wm[(NF + r) * NMSG + m];
            const float bmv = bm[m];
            for (int rep = 0; rep < 4; ++rep) {
                const int i = rep * 8 + (tid >> 5);
                float acc = 0.f;
                for (int j = 0; j < NA; ++j) {
                    float dij = sd[i][j];
                    if (i != j && dij <= CUTR) {
                        float s = shWm[j][m] + bmv;
#pragma unroll
                        for (int r = 0; r < NRBF; ++r)
                            s += (float)uA.E[i][j][r] * wme[r];
                        acc += s / (1.0f + __expf(-s));  // silu
                    }
                }
                sagg[i][m] = acc;
            }
        }
        __syncthreads();

        // h = h + silu(concat(h, agg) @ Wu + bu)
        for (int idx = tid; idx < NA * NF; idx += 256) {
            int a = idx >> 6, f = idx & 63;
            float s = bu[f];
#pragma unroll 8
            for (int c = 0; c < NF; ++c) s += sh[a][c] * Wu[c * NF + f];
#pragma unroll 8
            for (int c = 0; c < NMSG; ++c) s += sagg[a][c] * Wu[(NF + c) * NF + f];
            sh2[a][f] = sh[a][f] + s / (1.0f + __expf(-s));
        }
        __syncthreads();
        for (int idx = tid; idx < NA * NF; idx += 256)
            sh[idx >> 6][idx & 63] = sh2[idx >> 6][idx & 63];
        __syncthreads();
    }

    // -------------------- attention --------------------
    // tmp = h @ Wqkv + bqkv
    for (int idx = tid; idx < NA * 192; idx += 256) {
        int a = idx / 192, c = idx - a * 192;
        float s = bqkv[c];
#pragma unroll 8
        for (int f = 0; f < NF; ++f) s += sh[a][f] * Wqkv[f * 192 + c];
        tmp192[a][c] = (_Float16)s;
    }
    __syncthreads();   // E region dead from here on

    // q/k/v = split(tmp) @ {Wq,Wk,Wv} + bias
    for (int idx = tid; idx < NA * NF * 3; idx += 256) {
        int which = idx / (NA * NF);
        int a = (idx / NF) % NA;
        int f2 = idx & 63;
        const float* W  = (which == 0) ? Wq : (which == 1) ? Wk : Wv;
        const float* bb = (which == 0) ? bq : (which == 1) ? bk : bv;
        float s = bb[f2];
#pragma unroll 8
        for (int f = 0; f < NF; ++f)
            s += (float)tmp192[a][which * NF + f] * W[f * NF + f2];
        _Float16 val = (_Float16)s;
        if (which == 0) uA.att.q[a][f2] = val;
        else if (which == 1) uA.att.k[a][f2] = val;
        else uA.att.v[a][f2] = val;
    }
    __syncthreads();

    // softmax attention: one (head, query) per thread (8*32 == 256)
    {
        const int head = tid >> 5, qi = tid & 31;
        float sc[NA], mx = -1e30f;
#pragma unroll
        for (int kj = 0; kj < NA; ++kj) {
            float s = 0.f;
#pragma unroll
            for (int d = 0; d < 8; ++d)
                s += (float)uA.att.q[qi][head * 8 + d] * (float)uA.att.k[kj][head * 8 + d];
            s *= 0.3535533906f;   // 1/sqrt(8)
            sc[kj] = s;
            mx = fmaxf(mx, s);
        }
        float den = 0.f;
#pragma unroll
        for (int kj = 0; kj < NA; ++kj) { sc[kj] = __expf(sc[kj] - mx); den += sc[kj]; }
        const float inv = 1.0f / den;
#pragma unroll
        for (int d = 0; d < 8; ++d) {
            float o = 0.f;
#pragma unroll
            for (int kj = 0; kj < NA; ++kj)
                o += sc[kj] * (float)uA.att.v[kj][head * 8 + d];
            uA.att.o[qi][head * 8 + d] = (_Float16)(o * inv);
        }
    }
    __syncthreads();

    // sa = o @ Wo + bo ; h2 = LN1(h + sa)
    for (int idx = tid; idx < NA * NF; idx += 256) {
        int a = idx >> 6, f2 = idx & 63;
        float s = bo[f2];
#pragma unroll 8
        for (int f = 0; f < NF; ++f) s += (float)uA.att.o[a][f] * Wo[f * NF + f2];
        sh2[a][f2] = sh[a][f2] + s;
    }
    __syncthreads();
    if (tid < NA) {
        float mu = 0.f;
        for (int f = 0; f < NF; ++f) mu += sh2[tid][f];
        mu *= (1.0f / NF);
        float vv = 0.f;
        for (int f = 0; f < NF; ++f) { float t = sh2[tid][f] - mu; vv += t * t; }
        red[tid][0] = mu;
        red[tid][1] = rsqrtf(vv * (1.0f / NF) + 1e-5f);
    }
    __syncthreads();
    for (int idx = tid; idx < NA * NF; idx += 256) {
        int a = idx >> 6, f = idx & 63;
        sh2[a][f] = (sh2[a][f] - red[a][0]) * red[a][1] * g1[f] + be1[f];
    }
    __syncthreads();

    // FF: f1 = relu(h2 @ W1f + b1f)
    for (int idx = tid; idx < NA * NF; idx += 256) {
        int a = idx >> 6, c = idx & 63;
        float s = b1f[c];
#pragma unroll 8
        for (int f = 0; f < NF; ++f) s += sh2[a][f] * W1f[f * NF + c];
        tmp192[a][c] = (_Float16)fmaxf(s, 0.0f);
    }
    __syncthreads();
    // h3pre = h2 + f1 @ W2f + b2f
    for (int idx = tid; idx < NA * NF; idx += 256) {
        int a = idx >> 6, f2 = idx & 63;
        float s = b2f[f2];
#pragma unroll 8
        for (int c = 0; c < NF; ++c) s += (float)tmp192[a][c] * W2f[c * NF + f2];
        sh[a][f2] = sh2[a][f2] + s;
    }
    __syncthreads();
    if (tid < NA) {
        float mu = 0.f;
        for (int f = 0; f < NF; ++f) mu += sh[tid][f];
        mu *= (1.0f / NF);
        float vv = 0.f;
        for (int f = 0; f < NF; ++f) { float t = sh[tid][f] - mu; vv += t * t; }
        red[tid][0] = mu;
        red[tid][1] = rsqrtf(vv * (1.0f / NF) + 1e-5f);
    }
    __syncthreads();

    // Emit feat row (bf16, row-major, K = 2208)
    __bf16* fr = feat + (size_t)blockIdx.x * KFEAT;
    for (int idx = tid; idx < DREST; idx += 256) fr[idx] = (__bf16)xr[96 + idx];
    for (int idx = tid; idx < NA * NF; idx += 256) {
        int a = idx >> 6, f = idx & 63;
        float val = (sh[a][f] - red[a][0]) * red[a][1] * g2[f] + be2[f];
        fr[DREST + idx] = (__bf16)val;
    }
}

// ---------------------------------------------------------------------------
// Workspace layout (bytes, all 256-aligned)
// ---------------------------------------------------------------------------
static const size_t OFF_FEAT = 0;                                    // 4096*2208*2
static const size_t OFF_Y1   = OFF_FEAT + (size_t)BSZ * KFEAT * 2;   // 4096*1024*2
static const size_t OFF_Y2   = OFF_Y1   + (size_t)BSZ * HIDN * 2;
static const size_t OFF_W1P  = OFF_Y2   + (size_t)BSZ * HIDN * 2;    // 2208*1024*2
static const size_t OFF_W2P  = OFF_W1P  + (size_t)KFEAT * HIDN * 2;  // 1024*1024*2
static const size_t OFF_W3P  = OFF_W2P  + (size_t)HIDN * HIDN * 2;   // 1024*256*2

extern "C" void kernel_launch(void* const* d_in, const int* in_sizes, int n_in,
                              void* d_out, int out_size, void* d_ws, size_t ws_size,
                              hipStream_t stream) {
    const float* x    = (const float*)d_in[0];
    const float* ae   = (const float*)d_in[1];
    const float* Wm0  = (const float*)d_in[2];
    const float* bm0  = (const float*)d_in[3];
    const float* Wu0  = (const float*)d_in[4];
    const float* bu0  = (const float*)d_in[5];
    const float* Wm1  = (const float*)d_in[6];
    const float* bm1  = (const float*)d_in[7];
    const float* Wu1  = (const float*)d_in[8];
    const float* bu1  = (const float*)d_in[9];
    const float* Wqkv = (const float*)d_in[10];
    const float* bqkv = (const float*)d_in[11];
    const float* Wq   = (const float*)d_in[12];
    const float* Wk   = (const float*)d_in[13];
    const float* Wv   = (const float*)d_in[14];
    const float* Wo   = (const float*)d_in[15];
    const float* bq   = (const float*)d_in[16];
    const float* bk   = (const float*)d_in[17];
    const float* bv   = (const float*)d_in[18];
    const float* bo   = (const float*)d_in[19];
    const float* W1f  = (const float*)d_in[20];
    const float* b1f  = (const float*)d_in[21];
    const float* W2f  = (const float*)d_in[22];
    const float* b2f  = (const float*)d_in[23];
    const float* g1   = (const float*)d_in[24];
    const float* be1  = (const float*)d_in[25];
    const float* g2   = (const float*)d_in[26];
    const float* be2  = (const float*)d_in[27];
    const float* Wd1  = (const float*)d_in[28];
    const float* bd1  = (const float*)d_in[29];
    const float* Wd2  = (const float*)d_in[30];
    const float* bd2  = (const float*)d_in[31];
    const float* Wd3  = (const float*)d_in[32];
    const float* bd3  = (const float*)d_in[33];

    char* ws = (char*)d_ws;
    __bf16* feat = (__bf16*)(ws + OFF_FEAT);
    __bf16* y1   = (__bf16*)(ws + OFF_Y1);
    __bf16* y2   = (__bf16*)(ws + OFF_Y2);
    __bf16* w1p  = (__bf16*)(ws + OFF_W1P);
    __bf16* w2p  = (__bf16*)(ws + OFF_W2P);
    __bf16* w3p  = (__bf16*)(ws + OFF_W3P);

    // 1) pack the dense-head weights into WMMA B-fragment bf16 layout
    pack_b_kernel<<<(KFEAT * HIDN + 255) / 256, 256, 0, stream>>>(Wd1, w1p, KFEAT, HIDN);
    pack_b_kernel<<<(HIDN * HIDN + 255) / 256, 256, 0, stream>>>(Wd2, w2p, HIDN, HIDN);
    pack_b_kernel<<<(HIDN * DOUT + 255) / 256, 256, 0, stream>>>(Wd3, w3p, HIDN, DOUT);

    // 2) per-sample GNN front-end -> feat (bf16)
    gnn_kernel<<<BSZ, 256, 0, stream>>>(x, ae, Wm0, bm0, Wu0, bu0, Wm1, bm1, Wu1, bu1,
                                        Wqkv, bqkv, Wq, Wk, Wv, Wo, bq, bk, bv, bo,
                                        W1f, b1f, W2f, b2f, g1, be1, g2, be2, feat);

    // 3) dense head: three WMMA GEMMs (block tile 128x128)
    gemm_bf16_kernel<<<dim3(BSZ / 128, HIDN / 128), 256, 0, stream>>>(
        feat, w1p, bd1, (void*)y1, KFEAT, HIDN, 0);
    gemm_bf16_kernel<<<dim3(BSZ / 128, HIDN / 128), 256, 0, stream>>>(
        y1, w2p, bd2, (void*)y2, HIDN, HIDN, 0);
    gemm_bf16_kernel<<<dim3(BSZ / 128, DOUT / 128), 256, 0, stream>>>(
        y2, w3p, bd3, d_out, HIDN, DOUT, 1);
}
